// GATResNetBlock_17978733101322
// MI455X (gfx1250) — matrix operand order repaired and checked by reading, not compiled, not run on hardware
//
#include <hip/hip_runtime.h>

#define DIM 128
#define NEGS 0.2f
#define BNEPS 1e-5f

typedef __attribute__((ext_vector_type(16))) _Float16 v16h;
typedef __attribute__((ext_vector_type(8)))  float    v8f;

// ---------------------------------------------------------------------------
// Dual GEMM: XL = X @ Wl, XR = X @ Wr   (X: [M,128] f32, W: [128,128] f32)
// f16 WMMA, f32 accumulation. Weights staged in LDS pre-swizzled into WMMA
// B-fragment order so each lane fetches its fragment as one contiguous
// 32-byte LDS read (2x ds_load_b128) instead of 16 scalar ds_load_u16.
// One wave computes a 16-row strip across all 128 output columns.
// ---------------------------------------------------------------------------
__global__ __launch_bounds__(128) void gemm_dual_wmma(
    const float* __restrict__ X, const float* __restrict__ Wl,
    const float* __restrict__ Wr, float* __restrict__ XL,
    float* __restrict__ XR, int M)
{
  // Fragment-order layout: [t(4)][nt(8)][lane(32)][j(16)] halves = 16384
  __shared__ _Float16 sWl[DIM * DIM];
  __shared__ _Float16 sWr[DIM * DIM];
  const int tid = threadIdx.x;
  // B fragment (32x16 f16): lanes 0-15 hold K=kb+0..7 & kb+16..23 col n,
  // lanes 16-31 hold K=kb+8..15 & kb+24..31. For source element W[k][n]:
  //   t = k>>5, r = k&31, lane = (((r>>3)&1)<<4) | (n&15), nt = n>>4,
  //   j = (r&7) | ((r>>4)<<3)
  for (int sidx = tid; sidx < DIM * DIM; sidx += 128) {
    const int k = sidx >> 7;
    const int n = sidx & 127;
    const int t = k >> 5;
    const int r = k & 31;
    const int lane2 = (((r >> 3) & 1) << 4) | (n & 15);
    const int dst = ((((t << 3) | (n >> 4)) << 5 | lane2) << 4) | ((r & 7) | ((r >> 4) << 3));
    sWl[dst] = (_Float16)Wl[sidx];
    sWr[dst] = (_Float16)Wr[sidx];
  }
  __syncthreads();

  const int wave = tid >> 5;
  const int lane = tid & 31;
  const int m0   = (blockIdx.x * 4 + wave) * 16;
  if (m0 + 16 > M) return;

  const int lhalf = lane >> 4;   // 0 or 1
  const int lmod  = lane & 15;
  const float* xrow = X + (size_t)(m0 + lmod) * DIM;

  // A fragments (16x32 f16 layout): lanes 0-15 hold K=kk+0..7 & kk+16..23,
  // lanes 16-31 hold K=kk+8..15 & kk+24..31.
  v16h a[4];
#pragma unroll
  for (int t = 0; t < 4; ++t) {
    const int kb = t * 32 + lhalf * 8;
    float4 f0 = *(const float4*)(xrow + kb);
    float4 f1 = *(const float4*)(xrow + kb + 4);
    float4 f2 = *(const float4*)(xrow + kb + 16);
    float4 f3 = *(const float4*)(xrow + kb + 20);
    a[t][0]  = (_Float16)f0.x; a[t][1]  = (_Float16)f0.y;
    a[t][2]  = (_Float16)f0.z; a[t][3]  = (_Float16)f0.w;
    a[t][4]  = (_Float16)f1.x; a[t][5]  = (_Float16)f1.y;
    a[t][6]  = (_Float16)f1.z; a[t][7]  = (_Float16)f1.w;
    a[t][8]  = (_Float16)f2.x; a[t][9]  = (_Float16)f2.y;
    a[t][10] = (_Float16)f2.z; a[t][11] = (_Float16)f2.w;
    a[t][12] = (_Float16)f3.x; a[t][13] = (_Float16)f3.y;
    a[t][14] = (_Float16)f3.z; a[t][15] = (_Float16)f3.w;
  }

  for (int nt = 0; nt < 8; ++nt) {
    const int n = nt * 16 + lmod;
    v8f accL = {};
    v8f accR = {};
#pragma unroll
    for (int t = 0; t < 4; ++t) {
      const int foff = (((t << 3) | nt) << 5 | lane) << 4;
      v16h bl = *(const v16h*)(sWl + foff);
      v16h br = *(const v16h*)(sWr + foff);
      accL = __builtin_amdgcn_wmma_f32_16x16x32_f16(false, a[t], false, bl,
                                                    (short)0, accL, false, false);
      accR = __builtin_amdgcn_wmma_f32_16x16x32_f16(false, a[t], false, br,
                                                    (short)0, accR, false, false);
    }
    // C/D layout: vgpr r -> row m0 + r + 8*lhalf, col = nt*16 + lmod
#pragma unroll
    for (int r = 0; r < 8; ++r) {
      const size_t m = (size_t)(m0 + r + lhalf * 8);
      XL[m * DIM + n] = accL[r];
      XR[m * DIM + n] = accR[r];
    }
  }
}

// ---------------------------------------------------------------------------
// Ordered-uint encoding for float atomicMax
// ---------------------------------------------------------------------------
__device__ __forceinline__ unsigned f2ord(float v) {
  unsigned u = __float_as_uint(v);
  return (u & 0x80000000u) ? ~u : (u | 0x80000000u);
}
__device__ __forceinline__ float ord2f(unsigned k) {
  return (k & 0x80000000u) ? __uint_as_float(k & 0x7FFFFFFFu)
                           : __uint_as_float(~k);
}

// ---------------------------------------------------------------------------
// Edge kernels (wave32 per edge; lane owns 4 channels)
// ---------------------------------------------------------------------------
__global__ void edge_logits_kernel(
    const float* __restrict__ XL, const float* __restrict__ XR,
    const float* __restrict__ att, const long long* __restrict__ ei,
    int E, int Etot, float* __restrict__ elog, unsigned* __restrict__ emax)
{
  const long long gid = (long long)blockIdx.x * blockDim.x + threadIdx.x;
  const int e    = (int)(gid >> 5);
  const int lane = threadIdx.x & 31;
  if (e >= Etot) return;
  long long s, d;
  if (e < E) { s = ei[e]; d = ei[(size_t)E + e]; }
  else       { s = e - E; d = s; }

  float4 va = *(const float4*)(att + lane * 4);
  float4 vl = *(const float4*)(XL + (size_t)s * DIM + lane * 4);
  float4 vr = *(const float4*)(XR + (size_t)d * DIM + lane * 4);
  float h, p = 0.f;
  h = vl.x + vr.x; h = (h > 0.f) ? h : NEGS * h; p += va.x * h;
  h = vl.y + vr.y; h = (h > 0.f) ? h : NEGS * h; p += va.y * h;
  h = vl.z + vr.z; h = (h > 0.f) ? h : NEGS * h; p += va.z * h;
  h = vl.w + vr.w; h = (h > 0.f) ? h : NEGS * h; p += va.w * h;
#pragma unroll
  for (int off = 16; off > 0; off >>= 1) p += __shfl_down(p, off, 32);
  if (lane == 0) {
    elog[e] = p;
    atomicMax(&emax[d], f2ord(p));
  }
}

__global__ void edge_softmax_kernel(
    const float* __restrict__ elog, const unsigned* __restrict__ emax,
    const long long* __restrict__ ei, int E, int Etot,
    float* __restrict__ pbuf, float* __restrict__ denom)
{
  const int e = blockIdx.x * blockDim.x + threadIdx.x;
  if (e >= Etot) return;
  const long long d = (e < E) ? ei[(size_t)E + e] : (long long)(e - E);
  const unsigned k = emax[d];
  const float m = (k == 0u) ? 0.0f : ord2f(k);   // matches isfinite guard
  const float p = __expf(elog[e] - m);
  pbuf[e] = p;
  atomicAdd(&denom[d], p);
}

__global__ void edge_aggregate_kernel(
    const float* __restrict__ XL, const float* __restrict__ pbuf,
    const float* __restrict__ denom, const long long* __restrict__ ei,
    int E, int Etot, float* __restrict__ AGG)
{
  const long long gid = (long long)blockIdx.x * blockDim.x + threadIdx.x;
  const int e    = (int)(gid >> 5);
  const int lane = threadIdx.x & 31;
  if (e >= Etot) return;
  long long s, d;
  if (e < E) { s = ei[e]; d = ei[(size_t)E + e]; }
  else       { s = e - E; d = s; }
  const float alpha = pbuf[e] / fmaxf(denom[d], 1e-16f);
  float4 vl = *(const float4*)(XL + (size_t)s * DIM + lane * 4);
  float* o = AGG + (size_t)d * DIM + lane * 4;
  atomicAdd(o + 0, alpha * vl.x);
  atomicAdd(o + 1, alpha * vl.y);
  atomicAdd(o + 2, alpha * vl.z);
  atomicAdd(o + 3, alpha * vl.w);
}

// ---------------------------------------------------------------------------
// BatchNorm (training-mode) stats + apply kernels
// ---------------------------------------------------------------------------
__global__ __launch_bounds__(256) void bn_stats_kernel(
    const float* __restrict__ X, int n,
    float* __restrict__ mean, float* __restrict__ istd)
{
  const int c = blockIdx.x;
  float s = 0.f, q = 0.f;
  for (int i = threadIdx.x; i < n; i += 256) {
    const float v = X[(size_t)i * DIM + c];
    s += v; q += v * v;
  }
  __shared__ float rs[256], rq[256];
  rs[threadIdx.x] = s; rq[threadIdx.x] = q;
  __syncthreads();
  for (int off = 128; off > 0; off >>= 1) {
    if (threadIdx.x < off) {
      rs[threadIdx.x] += rs[threadIdx.x + off];
      rq[threadIdx.x] += rq[threadIdx.x + off];
    }
    __syncthreads();
  }
  if (threadIdx.x == 0) {
    const float mu  = rs[0] / n;
    const float var = rq[0] / n - mu * mu;
    mean[c] = mu;
    istd[c] = rsqrtf(fmaxf(var, 0.f) + BNEPS);
  }
}

__global__ void bn_apply_relu_kernel(
    const float* __restrict__ X, const float* __restrict__ mean,
    const float* __restrict__ istd, const float* __restrict__ g,
    const float* __restrict__ b, float* __restrict__ Y, int n4)
{
  const int i = blockIdx.x * blockDim.x + threadIdx.x;
  if (i >= n4) return;
  const int c = (i & 31) * 4;
  float4 v = ((const float4*)X)[i];
  v.x = fmaxf((v.x - mean[c + 0]) * istd[c + 0] * g[c + 0] + b[c + 0], 0.f);
  v.y = fmaxf((v.y - mean[c + 1]) * istd[c + 1] * g[c + 1] + b[c + 1], 0.f);
  v.z = fmaxf((v.z - mean[c + 2]) * istd[c + 2] * g[c + 2] + b[c + 2], 0.f);
  v.w = fmaxf((v.w - mean[c + 3]) * istd[c + 3] * g[c + 3] + b[c + 3], 0.f);
  ((float4*)Y)[i] = v;
}

__global__ void bn_add_relu_kernel(
    const float* __restrict__ A, const float* __restrict__ B,
    const float* __restrict__ mA, const float* __restrict__ iA,
    const float* __restrict__ gA, const float* __restrict__ bA,
    const float* __restrict__ mB, const float* __restrict__ iB,
    const float* __restrict__ gB, const float* __restrict__ bB,
    float* __restrict__ out, int n4)
{
  const int i = blockIdx.x * blockDim.x + threadIdx.x;
  if (i >= n4) return;
  const int c = (i & 31) * 4;
  float4 va = ((const float4*)A)[i];
  float4 vb = ((const float4*)B)[i];
  float4 r;
  r.x = fmaxf((va.x - mA[c+0]) * iA[c+0] * gA[c+0] + bA[c+0] +
              (vb.x - mB[c+0]) * iB[c+0] * gB[c+0] + bB[c+0], 0.f);
  r.y = fmaxf((va.y - mA[c+1]) * iA[c+1] * gA[c+1] + bA[c+1] +
              (vb.y - mB[c+1]) * iB[c+1] * gB[c+1] + bB[c+1], 0.f);
  r.z = fmaxf((va.z - mA[c+2]) * iA[c+2] * gA[c+2] + bA[c+2] +
              (vb.z - mB[c+2]) * iB[c+2] * gB[c+2] + bB[c+2], 0.f);
  r.w = fmaxf((va.w - mA[c+3]) * iA[c+3] * gA[c+3] + bA[c+3] +
              (vb.w - mB[c+3]) * iB[c+3] * gB[c+3] + bB[c+3], 0.f);
  ((float4*)out)[i] = r;
}

__global__ void fill_zero4_kernel(float4* __restrict__ p, int n4) {
  const int i = blockIdx.x * blockDim.x + threadIdx.x;
  if (i < n4) p[i] = make_float4(0.f, 0.f, 0.f, 0.f);
}

// ---------------------------------------------------------------------------
// Host orchestration
// ---------------------------------------------------------------------------
extern "C" void kernel_launch(void* const* d_in, const int* in_sizes, int n_in,
                              void* d_out, int out_size, void* d_ws, size_t ws_size,
                              hipStream_t stream)
{
  (void)n_in; (void)out_size; (void)ws_size;
  const float*     x      = (const float*)d_in[0];
  const long long* ei     = (const long long*)d_in[1];  // int64 edge_index [2,E]
  const float* g1_wl = (const float*)d_in[2];
  const float* g1_wr = (const float*)d_in[3];
  const float* g1_at = (const float*)d_in[4];
  // d_in[5] g1_b: constant per-channel shift, cancels exactly in BatchNorm
  const float* bn1_g = (const float*)d_in[6];
  const float* bn1_b = (const float*)d_in[7];
  const float* g2_wl = (const float*)d_in[8];
  const float* g2_wr = (const float*)d_in[9];
  const float* g2_at = (const float*)d_in[10];
  const float* bn2_g = (const float*)d_in[12];
  const float* bn2_b = (const float*)d_in[13];
  const float* gs_wl = (const float*)d_in[14];
  const float* gs_wr = (const float*)d_in[15];
  const float* gs_at = (const float*)d_in[16];
  const float* bn3_g = (const float*)d_in[18];
  const float* bn3_b = (const float*)d_in[19];

  const int N    = in_sizes[0] / DIM;   // 50000
  const int E    = in_sizes[1] / 2;     // 800000
  const int Etot = E + N;               // + self loops
  const size_t NB = (size_t)N * DIM;

  char* w = (char*)d_ws;
  auto alloc = [&](size_t bytes) -> char* {
    char* p = w; w += (bytes + 255) & ~(size_t)255; return p;
  };
  float*    XL    = (float*)alloc(NB * 4);
  float*    XR    = (float*)alloc(NB * 4);
  float*    AGGA  = (float*)alloc(NB * 4);
  float*    AGGS  = (float*)alloc(NB * 4);
  float*    H     = (float*)alloc(NB * 4);
  float*    ELOG  = (float*)alloc((size_t)Etot * 4);
  float*    PBUF  = (float*)alloc((size_t)Etot * 4);
  unsigned* EMAX  = (unsigned*)alloc((size_t)N * 4);
  float*    DEN   = (float*)alloc((size_t)N * 4);
  float* MEAN1 = (float*)alloc(DIM * 4); float* ISTD1 = (float*)alloc(DIM * 4);
  float* MEAN2 = (float*)alloc(DIM * 4); float* ISTD2 = (float*)alloc(DIM * 4);
  float* MEAN3 = (float*)alloc(DIM * 4); float* ISTD3 = (float*)alloc(DIM * 4);

  const int mtiles  = (N + 15) / 16;
  const int gblocks = (mtiles + 3) / 4;
  const int lblocks = (int)(((long long)Etot * 32 + 255) / 256);
  const int sblocks = (Etot + 255) / 256;
  const int nb4 = (int)(NB / 4);
  const int fb  = (nb4 + 255) / 256;
  const int n4  = N / 4;
  const int fnb = (n4 + 255) / 256;

  auto edge_pass = [&](const float* xl, const float* xr, const float* att,
                       float* agg) {
    fill_zero4_kernel<<<fb,  256, 0, stream>>>((float4*)agg, nb4);
    fill_zero4_kernel<<<fnb, 256, 0, stream>>>((float4*)EMAX, n4);
    fill_zero4_kernel<<<fnb, 256, 0, stream>>>((float4*)DEN, n4);
    edge_logits_kernel<<<lblocks, 256, 0, stream>>>(xl, xr, att, ei, E, Etot,
                                                    ELOG, EMAX);
    edge_softmax_kernel<<<sblocks, 256, 0, stream>>>(ELOG, EMAX, ei, E, Etot,
                                                     PBUF, DEN);
    edge_aggregate_kernel<<<lblocks, 256, 0, stream>>>(xl, PBUF, DEN, ei, E,
                                                       Etot, agg);
  };

  // --- GAT layer 1 + BN1 + ReLU -------------------------------------------
  gemm_dual_wmma<<<gblocks, 128, 0, stream>>>(x, g1_wl, g1_wr, XL, XR, N);
  edge_pass(XL, XR, g1_at, AGGA);
  bn_stats_kernel<<<DIM, 256, 0, stream>>>(AGGA, N, MEAN1, ISTD1);
  bn_apply_relu_kernel<<<fb, 256, 0, stream>>>(AGGA, MEAN1, ISTD1, bn1_g,
                                               bn1_b, H, nb4);
  // --- GAT layer 2 ---------------------------------------------------------
  gemm_dual_wmma<<<gblocks, 128, 0, stream>>>(H, g2_wl, g2_wr, XL, XR, N);
  edge_pass(XL, XR, g2_at, AGGA);
  // --- Skip GAT layer ------------------------------------------------------
  gemm_dual_wmma<<<gblocks, 128, 0, stream>>>(x, gs_wl, gs_wr, XL, XR, N);
  edge_pass(XL, XR, gs_at, AGGS);
  // --- BN2 + BN3 + add + ReLU ---------------------------------------------
  bn_stats_kernel<<<DIM, 256, 0, stream>>>(AGGA, N, MEAN2, ISTD2);
  bn_stats_kernel<<<DIM, 256, 0, stream>>>(AGGS, N, MEAN3, ISTD3);
  bn_add_relu_kernel<<<fb, 256, 0, stream>>>(AGGA, AGGS, MEAN2, ISTD2, bn2_g,
                                             bn2_b, MEAN3, ISTD3, bn3_g, bn3_b,
                                             (float*)d_out, nb4);
}
